// BlockAttention_74483322847913
// MI455X (gfx1250) — matrix operand order repaired and checked by reading, not compiled, not run on hardware
//
#include <hip/hip_runtime.h>
#include <hip/hip_bf16.h>
#include <math.h>

// ---------------------------------------------------------------------------
// CDNA5 (gfx1250) fused windowed-attention block.
// bf16 WMMA (v_wmma_f32_16x16x32_bf16), f32 accumulate, wave32.
// ---------------------------------------------------------------------------

typedef __attribute__((ext_vector_type(16))) __bf16 bf16x16;
typedef __attribute__((ext_vector_type(8)))  __bf16 bf16x8;
typedef __attribute__((ext_vector_type(8)))  float  f32x8;

#define DIM     256
#define HID     1024
#define NTOK    64          // tokens per 8x8 window
#define HEADS   8
#define DH      32
#define NWIN    2048        // 2 * 32 * 32 windows
#define HW      256         // H == W

// LDS row strides (padded to break 64-bank conflicts; keep 16B alignment)
#define XSTR    260         // f32 rows (64 x 260 f32)
#define TSTR    264         // bf16 rows, 264*2 = 528B, 16B aligned
#define KSTR    72          // kT / p rows (bf16), 144B
#define PSTR    72

// ---- WMMA fragment loaders -------------------------------------------------
// A: row-major MxK. lane(l&15)=row M. lanes<16 hold K {0..7,16..23}+kbase,
// lanes>=16 hold K {8..15,24..31}+kbase (ISA 7.12.2, 16-bit A 16x32).
__device__ __forceinline__ bf16x16 load_frag_a(const __bf16* base, int stride, int kbase) {
    const int l = threadIdx.x & 31;
    const __bf16* p = base + (l & 15) * stride + kbase + ((l >> 4) << 3);
    bf16x8 lo = *(const bf16x8*)(p);
    bf16x8 hi = *(const bf16x8*)(p + 16);
    bf16x16 r;
#pragma unroll
    for (int i = 0; i < 8; ++i) { r[i] = lo[i]; r[i + 8] = hi[i]; }
    return r;
}

// B: row-major KxN. lane(l&15)=col N+nbase. lanes<16 hold K {0..7,16..23}+kbase,
// lanes>=16 hold K {8..15,24..31}+kbase (same K order as A).
__device__ __forceinline__ bf16x16 load_frag_b(const __bf16* base, int stride,
                                               int kbase, int nbase) {
    const int l = threadIdx.x & 31;
    const int kr = kbase + ((l >> 4) << 3);
    const __bf16* p = base + kr * stride + nbase + (l & 15);
    bf16x16 r;
#pragma unroll
    for (int i = 0; i < 8; ++i) { r[i] = p[i * stride]; r[i + 8] = p[(16 + i) * stride]; }
    return r;
}

__device__ __forceinline__ f32x8 wmma_bf16(bf16x16 a, bf16x16 b, f32x8 c) {
    return __builtin_amdgcn_wmma_f32_16x16x32_bf16(false, a, false, b,
                                                   (short)0, c, false, false);
}

// 4-lane partial-sum reduce (tokens owned by lane groups of 4)
__device__ __forceinline__ void reduce4(float& s, float& ss) {
    s  += __shfl_xor(s, 1, 32);  ss += __shfl_xor(ss, 1, 32);
    s  += __shfl_xor(s, 2, 32);  ss += __shfl_xor(ss, 2, 32);
}

// ---- tiny prep kernels -----------------------------------------------------
__global__ void cvt_bf16_kernel(const float* __restrict__ in, __bf16* __restrict__ out, int n) {
    int i = blockIdx.x * 256 + threadIdx.x;
    if (i < n) out[i] = (__bf16)in[i];
}

// bias[h][i][j] = rpb[rel_pos_idx(i,j)][h], 8x64x64 f32
__global__ void build_bias_kernel(const float* __restrict__ rpb, float* __restrict__ bias) {
    int idx = blockIdx.x * 256 + threadIdx.x;          // 32768 total
    int h = idx >> 12, i = (idx >> 6) & 63, j = idx & 63;
    int yi = i >> 3, xi = i & 7, yj = j >> 3, xj = j & 7;
    int rel = (yi - yj + 7) * 15 + (xi - xj + 7);
    bias[idx] = rpb[rel * HEADS + h];
}

// ---- attention kernel: one window per block (256 thr = 8 waves) ------------
// LDS layout (dynamic, 278528 B):
//   xw  f32 [64][260]   raw window (residual source)        66560 B
//   t   bf16[64][264]   LN2(LN1(x))  -> reused as o         33792 B
//   q   bf16[64][264]   scaled q                            33792 B
//   kT  bf16[256][72]   k transposed (feature-major)        36864 B
//   v   bf16[64][264]                                       33792 B
//   p   bf16[8][64][72] softmax probs per head              73728 B
#define ATTN_LDS (66560 + 33792 + 33792 + 36864 + 33792 + 73728)

__global__ __launch_bounds__(256, 1)
void attn_kernel(const float* __restrict__ x,
                 const float* __restrict__ ln1g, const float* __restrict__ ln1b,
                 const float* __restrict__ ln2g, const float* __restrict__ ln2b,
                 const __bf16* __restrict__ wqkv, const __bf16* __restrict__ wout,
                 const float* __restrict__ bias, float* __restrict__ xw2) {
    extern __shared__ char smem[];
    float*  xw = (float*)smem;
    __bf16* t  = (__bf16*)(smem + 66560);
    __bf16* q  = t + NTOK * TSTR;
    __bf16* kT = q + NTOK * TSTR;
    __bf16* v  = kT + DIM * KSTR;
    __bf16* p  = v + NTOK * TSTR;
    __bf16* o  = t;                       // reuse after QKV phase
    __shared__ float stats[4 * 64];

    const int tid = threadIdx.x;
    // wave id, pinned to an SGPR so region branches become scalar s_cbranch
    const int wv  = __builtin_amdgcn_readfirstlane(tid >> 5);
    const int ln  = tid & 31;
    const int hf  = ln >> 4;              // D-layout half (M offset 8)
    const int nn  = ln & 15;              // D-layout N lane

    const int win = blockIdx.x;
    const int bb  = win >> 10;
    const int rem = win & 1023;
    const int h0  = (rem >> 5) << 3;      // window top row
    const int w0  = (rem & 31) << 3;      // window left col

    if (tid < 8) {  // keep weights warm in L2 across 2048 windows
        __builtin_prefetch(wqkv + tid * (DIM * 3 * DIM / 8), 0, 1);
        __builtin_prefetch(wout + tid * (DIM * DIM / 8), 0, 1);
    }

    // -- gather NCHW window -> xw (32B global segments, LDS transpose) --
    {
        const int cc = tid & 7, r = (tid >> 3) & 7, clo = tid >> 6;
        const float* xb = x + (size_t)bb * DIM * HW * HW + (size_t)h0 * HW + w0;
#pragma unroll 4
        for (int ch = 0; ch < 64; ++ch) {
            int c = ch * 4 + clo;
            xw[(r * 8 + cc) * XSTR + c] = xb[(size_t)c * (HW * HW) + r * HW + cc];
        }
    }
    __syncthreads();

    // -- double LayerNorm (LN1 then LN2): 4 threads per token --
    {
        const int tk = tid >> 2, qd = tid & 3, qb = qd << 6;
        float s = 0.f, ss = 0.f;
        for (int j = 0; j < 64; ++j) {                 // bank-staggered
            int c = qb + ((j + (qd << 4)) & 63);
            float vv = xw[tk * XSTR + c];
            s += vv; ss += vv * vv;
        }
        reduce4(s, ss);
        const float m1 = s * (1.f / DIM);
        const float r1 = rsqrtf(ss * (1.f / DIM) - m1 * m1 + 1e-5f);
        float s2 = 0.f, ss2 = 0.f;
        for (int j = 0; j < 64; ++j) {
            int c = qb + ((j + (qd << 4)) & 63);
            float y1 = (xw[tk * XSTR + c] - m1) * r1 * ln1g[c] + ln1b[c];
            s2 += y1; ss2 += y1 * y1;
        }
        reduce4(s2, ss2);
        const float m2 = s2 * (1.f / DIM);
        const float r2 = rsqrtf(ss2 * (1.f / DIM) - m2 * m2 + 1e-5f);
        if (qd == 0) {
            stats[tk] = m1; stats[64 + tk] = r1;
            stats[128 + tk] = m2; stats[192 + tk] = r2;
        }
    }
    __syncthreads();
    for (int idx = tid; idx < NTOK * DIM; idx += 256) {
        int tk = idx >> 8, c = idx & 255;
        float y1 = (xw[tk * XSTR + c] - stats[tk]) * stats[64 + tk] * ln1g[c] + ln1b[c];
        float y2 = (y1 - stats[128 + tk]) * stats[192 + tk] * ln2g[c] + ln2b[c];
        t[tk * TSTR + c] = (__bf16)y2;
    }
    __syncthreads();

    // -- QKV GEMM: (64x256)x(256x768). wave wv owns N in [wv*96, wv*96+96) --
#pragma unroll
    for (int mt = 0; mt < 4; ++mt) {
        bf16x16 af[8];
#pragma unroll
        for (int kk = 0; kk < 8; ++kk) af[kk] = load_frag_a(t + mt * 16 * TSTR, TSTR, kk * 32);
        for (int nt = 0; nt < 6; ++nt) {
            const int nb = wv * 96 + nt * 16;          // SGPR; tile never straddles q/k/v
            f32x8 acc = {};
#pragma unroll
            for (int kk = 0; kk < 8; ++kk)
                acc = wmma_bf16(af[kk], load_frag_b(wqkv, 3 * DIM, kk * 32, nb), acc);
            const int mbase = mt * 16 + hf * 8;
            if (nb < 256) {                            // scalar branch (nb uniform)
#pragma unroll
                for (int r2 = 0; r2 < 8; ++r2)
                    q[(mbase + r2) * TSTR + nb + nn] = (__bf16)(acc[r2] * 0.17677669529663687f);
            } else if (nb < 512) {                     // packed b128 store (consecutive m)
                bf16x8 pk;
#pragma unroll
                for (int r2 = 0; r2 < 8; ++r2) pk[r2] = (__bf16)acc[r2];
                *(bf16x8*)(kT + (nb - 256 + nn) * KSTR + mbase) = pk;
            } else {
#pragma unroll
                for (int r2 = 0; r2 < 8; ++r2)
                    v[(mbase + r2) * TSTR + (nb - 512 + nn)] = (__bf16)acc[r2];
            }
        }
    }
    __syncthreads();

    // -- attention core: wave wv == head h, full 64x64 scores in registers --
    const int h = wv;                                   // SGPR
    f32x8 s[4][4];
    {
        bf16x16 bk[4];
#pragma unroll
        for (int nt = 0; nt < 4; ++nt)
            bk[nt] = load_frag_b(kT + h * DH * KSTR, KSTR, 0, nt * 16);
#pragma unroll
        for (int mt = 0; mt < 4; ++mt) {
            bf16x16 aq = load_frag_a(q + mt * 16 * TSTR + h * DH, TSTR, 0); // K = dh = 32
#pragma unroll
            for (int nt = 0; nt < 4; ++nt) {
                f32x8 acc = {};
                s[mt][nt] = wmma_bf16(aq, bk[nt], acc);
            }
        }
    }
    // bias + row softmax (rows live in D-layout: lane-half = M offset 8)
#pragma unroll
    for (int mt = 0; mt < 4; ++mt)
#pragma unroll
        for (int nt = 0; nt < 4; ++nt)
#pragma unroll
            for (int r2 = 0; r2 < 8; ++r2)
                s[mt][nt][r2] += bias[h * 4096 + (mt * 16 + r2 + hf * 8) * 64 + nt * 16 + nn];
#pragma unroll
    for (int mt = 0; mt < 4; ++mt) {
#pragma unroll
        for (int r2 = 0; r2 < 8; ++r2) {
            float mx = fmaxf(fmaxf(s[mt][0][r2], s[mt][1][r2]),
                             fmaxf(s[mt][2][r2], s[mt][3][r2]));
            for (int off = 8; off > 0; off >>= 1) mx = fmaxf(mx, __shfl_xor(mx, off, 32));
            float sum = 0.f;
#pragma unroll
            for (int nt = 0; nt < 4; ++nt) {
                float e = __expf(s[mt][nt][r2] - mx);
                s[mt][nt][r2] = e; sum += e;
            }
            for (int off = 8; off > 0; off >>= 1) sum += __shfl_xor(sum, off, 32);
            const float inv = 1.f / sum;
#pragma unroll
            for (int nt = 0; nt < 4; ++nt)
                p[h * 64 * PSTR + (mt * 16 + r2 + hf * 8) * PSTR + nt * 16 + nn]
                    = (__bf16)(s[mt][nt][r2] * inv);
        }
    }
    // P @ V  (64x64 x 64x32), same wave: DS in-order, no barrier needed
    {
        bf16x16 bv[2][2];
#pragma unroll
        for (int nt = 0; nt < 2; ++nt)
#pragma unroll
            for (int kk = 0; kk < 2; ++kk)
                bv[nt][kk] = load_frag_b(v, TSTR, kk * 32, h * DH + nt * 16);
#pragma unroll
        for (int mt = 0; mt < 4; ++mt) {
            bf16x16 ap[2];
#pragma unroll
            for (int kk = 0; kk < 2; ++kk)
                ap[kk] = load_frag_a(p + h * 64 * PSTR + mt * 16 * PSTR, PSTR, kk * 32);
#pragma unroll
            for (int nt = 0; nt < 2; ++nt) {
                f32x8 acc = {};
#pragma unroll
                for (int kk = 0; kk < 2; ++kk) acc = wmma_bf16(ap[kk], bv[nt][kk], acc);
#pragma unroll
                for (int r2 = 0; r2 < 8; ++r2)
                    o[(mt * 16 + r2 + hf * 8) * TSTR + h * DH + nt * 16 + nn] = (__bf16)acc[r2];
            }
        }
    }
    __syncthreads();

    // -- out = o @ Wout + xw residual; wave wv owns N in [wv*32, wv*32+32) --
#pragma unroll
    for (int mt = 0; mt < 4; ++mt) {
        bf16x16 af[8];
#pragma unroll
        for (int kk = 0; kk < 8; ++kk) af[kk] = load_frag_a(o + mt * 16 * TSTR, TSTR, kk * 32);
#pragma unroll
        for (int nt = 0; nt < 2; ++nt) {
            const int nb = wv * 32 + nt * 16;           // SGPR
            f32x8 acc = {};
#pragma unroll
            for (int kk = 0; kk < 8; ++kk)
                acc = wmma_bf16(af[kk], load_frag_b(wout, DIM, kk * 32, nb), acc);
            const int c = nb + nn;
#pragma unroll
            for (int r2 = 0; r2 < 8; ++r2) {
                const int m = mt * 16 + r2 + hf * 8;
                xw2[((size_t)win * NTOK + m) * DIM + c] = acc[r2] + xw[m * XSTR + c];
            }
        }
    }
}

// ---- FFN kernel: LN3 -> x@W1 + b1 -> exact GELU -> @W2 + b2 -> residual ----
// LDS: xt f32[64][260] 66560 | a bf16[64][264] 33792 | hh bf16[64][264] 33792
#define FFN_LDS (66560 + 33792 + 33792)

__global__ __launch_bounds__(256, 1)
void ffn_kernel(const float* __restrict__ xw2,
                const float* __restrict__ ln3g, const float* __restrict__ ln3b,
                const __bf16* __restrict__ w1, const float* __restrict__ b1,
                const __bf16* __restrict__ w2, const float* __restrict__ b2,
                float* __restrict__ out) {
    extern __shared__ char smem[];
    float*  xt = (float*)smem;
    __bf16* a  = (__bf16*)(smem + 66560);
    __bf16* hh = a + NTOK * TSTR;
    __shared__ float stats[128];

    const int tid = threadIdx.x;
    const int wv  = __builtin_amdgcn_readfirstlane(tid >> 5);   // SGPR wave id
    const int ln  = tid & 31;
    const int hf  = ln >> 4;
    const int nn  = ln & 15;

    const int win = blockIdx.x;
    const int bb  = win >> 10;
    const int rem = win & 1023;
    const int h0  = (rem >> 5) << 3;
    const int w0  = (rem & 31) << 3;

    // load token tile (fully coalesced)
    for (int idx = tid; idx < NTOK * DIM; idx += 256) {
        int tk = idx >> 8, c = idx & 255;
        xt[tk * XSTR + c] = xw2[(size_t)win * (NTOK * DIM) + idx];
    }
    __syncthreads();
    // LN3 stats: 4 threads per token
    {
        const int tk = tid >> 2, qd = tid & 3, qb = qd << 6;
        float s = 0.f, ss = 0.f;
        for (int j = 0; j < 64; ++j) {
            int c = qb + ((j + (qd << 4)) & 63);
            float vv = xt[tk * XSTR + c];
            s += vv; ss += vv * vv;
        }
        reduce4(s, ss);
        if (qd == 0) {
            float m = s * (1.f / DIM);
            stats[tk] = m;
            stats[64 + tk] = rsqrtf(ss * (1.f / DIM) - m * m + 1e-5f);
        }
    }
    __syncthreads();
    for (int idx = tid; idx < NTOK * DIM; idx += 256) {
        int tk = idx >> 8, c = idx & 255;
        a[tk * TSTR + c] = (__bf16)((xt[tk * XSTR + c] - stats[tk]) * stats[64 + tk] * ln3g[c] + ln3b[c]);
    }
    __syncthreads();

    f32x8 oacc[4][2];
#pragma unroll
    for (int mt = 0; mt < 4; ++mt)
#pragma unroll
        for (int nt = 0; nt < 2; ++nt) oacc[mt][nt] = (f32x8){};

    for (int ch = 0; ch < 4; ++ch) {           // hidden in 4 chunks of 256
        // h_chunk = GELU(a @ W1[:, ch*256 + wv*32 ...] + b1)
#pragma unroll
        for (int mt = 0; mt < 4; ++mt) {
            bf16x16 af[8];
#pragma unroll
            for (int kk = 0; kk < 8; ++kk) af[kk] = load_frag_a(a + mt * 16 * TSTR, TSTR, kk * 32);
#pragma unroll
            for (int nt = 0; nt < 2; ++nt) {
                const int nb = ch * 256 + wv * 32 + nt * 16;    // SGPR
                f32x8 acc = {};
#pragma unroll
                for (int kk = 0; kk < 8; ++kk)
                    acc = wmma_bf16(af[kk], load_frag_b(w1, HID, kk * 32, nb), acc);
                const float bv = b1[nb + nn];
#pragma unroll
                for (int r2 = 0; r2 < 8; ++r2) {
                    float vv = acc[r2] + bv;
                    vv = 0.5f * vv * (1.f + erff(vv * 0.70710678118654752f)); // exact GELU
                    hh[(mt * 16 + r2 + hf * 8) * TSTR + wv * 32 + nt * 16 + nn] = (__bf16)vv;
                }
            }
        }
        __syncthreads();
        // out += h_chunk @ W2[ch*256 : ch*256+256, :]
#pragma unroll
        for (int mt = 0; mt < 4; ++mt) {
            bf16x16 af[8];
#pragma unroll
            for (int kk = 0; kk < 8; ++kk) af[kk] = load_frag_a(hh + mt * 16 * TSTR, TSTR, kk * 32);
#pragma unroll
            for (int nt = 0; nt < 2; ++nt) {
                const int nb = wv * 32 + nt * 16;               // SGPR
#pragma unroll
                for (int kk = 0; kk < 8; ++kk)
                    oacc[mt][nt] = wmma_bf16(af[kk],
                        load_frag_b(w2 + (size_t)ch * 256 * DIM, DIM, kk * 32, nb), oacc[mt][nt]);
            }
        }
        __syncthreads();
    }

    // residual + b2, stage final f32 tile back into xt
#pragma unroll
    for (int mt = 0; mt < 4; ++mt)
#pragma unroll
        for (int nt = 0; nt < 2; ++nt) {
            const int c = wv * 32 + nt * 16 + nn;
            const float bv = b2[c];
#pragma unroll
            for (int r2 = 0; r2 < 8; ++r2) {
                const int m = mt * 16 + r2 + hf * 8;
                xt[m * XSTR + c] = oacc[mt][nt][r2] + bv + xt[m * XSTR + c];
            }
        }
    __syncthreads();

    // scatter back to NCHW (8-float = 32B runs along W)
    {
        const int cc = tid & 7, r = (tid >> 3) & 7, clo = tid >> 6;
        float* ob = out + (size_t)bb * DIM * HW * HW + (size_t)h0 * HW + w0;
#pragma unroll 4
        for (int ch = 0; ch < 64; ++ch) {
            int c = ch * 4 + clo;
            ob[(size_t)c * (HW * HW) + r * HW + cc] = xt[(r * 8 + cc) * XSTR + c];
        }
    }
}

// ---------------------------------------------------------------------------
extern "C" void kernel_launch(void* const* d_in, const int* in_sizes, int n_in,
                              void* d_out, int out_size, void* d_ws, size_t ws_size,
                              hipStream_t stream) {
    const float* x     = (const float*)d_in[0];
    const float* ln1g  = (const float*)d_in[1];
    const float* ln1b  = (const float*)d_in[2];
    const float* ln2g  = (const float*)d_in[3];
    const float* ln2b  = (const float*)d_in[4];
    const float* w_qkv = (const float*)d_in[5];
    const float* w_out = (const float*)d_in[6];
    const float* rpb   = (const float*)d_in[7];
    const float* ln3g  = (const float*)d_in[8];
    const float* ln3b  = (const float*)d_in[9];
    const float* w1    = (const float*)d_in[10];
    const float* b1    = (const float*)d_in[11];
    const float* w2    = (const float*)d_in[12];
    const float* b2    = (const float*)d_in[13];
    float* outp = (float*)d_out;

    // workspace carve-up (all 256B-aligned)
    char* ws = (char*)d_ws;
    __bf16* wq_bf = (__bf16*)ws;                     ws += (size_t)DIM * 3 * DIM * 2;   // 384KB
    __bf16* wo_bf = (__bf16*)ws;                     ws += (size_t)DIM * DIM * 2;       // 128KB
    __bf16* w1_bf = (__bf16*)ws;                     ws += (size_t)DIM * HID * 2;       // 512KB
    __bf16* w2_bf = (__bf16*)ws;                     ws += (size_t)HID * DIM * 2;       // 512KB
    float*  biasg = (float*)ws;                      ws += (size_t)HEADS * 64 * 64 * 4; // 128KB
    float*  xw2   = (float*)ws;                      // 2048*64*256*4 = 128MB

    cvt_bf16_kernel<<<(DIM * 3 * DIM + 255) / 256, 256, 0, stream>>>(w_qkv, wq_bf, DIM * 3 * DIM);
    cvt_bf16_kernel<<<(DIM * DIM + 255) / 256, 256, 0, stream>>>(w_out, wo_bf, DIM * DIM);
    cvt_bf16_kernel<<<(DIM * HID + 255) / 256, 256, 0, stream>>>(w1, w1_bf, DIM * HID);
    cvt_bf16_kernel<<<(HID * DIM + 255) / 256, 256, 0, stream>>>(w2, w2_bf, HID * DIM);
    build_bias_kernel<<<(HEADS * 64 * 64) / 256, 256, 0, stream>>>(rpb, biasg);

    static bool attr_done = []() {
        hipFuncSetAttribute((const void*)attn_kernel,
                            hipFuncAttributeMaxDynamicSharedMemorySize, ATTN_LDS);
        hipFuncSetAttribute((const void*)ffn_kernel,
                            hipFuncAttributeMaxDynamicSharedMemorySize, FFN_LDS);
        return true;
    }();
    (void)attr_done;

    attn_kernel<<<NWIN, 256, ATTN_LDS, stream>>>(x, ln1g, ln1b, ln2g, ln2b,
                                                 wq_bf, wo_bf, biasg, xw2);
    ffn_kernel<<<NWIN, 256, FFN_LDS, stream>>>(xw2, ln3g, ln3b,
                                               w1_bf, b1, w2_bf, b2, outp);
}
